// _PagedMiniCPM4Attention_76811195122118
// MI455X (gfx1250) — matrix operand order repaired and checked by reading, not compiled, not run on hardware
//
#include <hip/hip_runtime.h>
#include <hip/hip_bf16.h>

#define T_TOK 2048
#define HIDN  2048
#define NH    16
#define NKV   2
#define HD    128
#define QKV_N (NH*HD + 2*NKV*HD)   // 2560
#define GROUP (NH/NKV)             // 8

typedef __attribute__((ext_vector_type(16))) __bf16 v16bf;
typedef __attribute__((ext_vector_type(8)))  __bf16 v8bf;
typedef __attribute__((ext_vector_type(8)))  float  v8f;

// ---------------------------------------------------------------------------
// Async global -> LDS copy of one 16-byte chunk (CDNA5, tracked by ASYNCcnt).
// ---------------------------------------------------------------------------
static __device__ inline void async_copy_b128(unsigned lds_off, uint64_t gaddr) {
    asm volatile("global_load_async_to_lds_b128 %0, %1, off"
                 :: "v"(lds_off), "v"(gaddr) : "memory");
}

// A 16x32 bf16 fragment from an LDS tile with 32-element rows.
// lanes 0-15 hold M=lane, K in {0..7,16..23}; lanes 16-31 M=lane-16,
// K in {8..15,24..31}.
static __device__ inline v16bf a_frag_lds(const __bf16* tile, int row0, int lane) {
    int m  = lane & 15;
    int kb = (lane >> 4) << 3;
    const __bf16* p = tile + (row0 + m) * 32 + kb;
    v8bf lo = *(const v8bf*)(p);
    v8bf hi = *(const v8bf*)(p + 16);
    v16bf a;
#pragma unroll
    for (int i = 0; i < 8; ++i) { a[i] = lo[i]; a[8 + i] = hi[i]; }
    return a;
}

// A 16x32 bf16 fragment straight from global (row-major, stride lda).
static __device__ inline v16bf a_frag_global(const __bf16* __restrict__ base,
                                             int lda, int row0, int k0, int lane) {
    int m  = lane & 15;
    int kb = (lane >> 4) << 3;
    const __bf16* p = base + (size_t)(row0 + m) * lda + k0 + kb;
    v8bf lo = *(const v8bf*)(p);
    v8bf hi = *(const v8bf*)(p + 16);
    v16bf a;
#pragma unroll
    for (int i = 0; i < 8; ++i) { a[i] = lo[i]; a[8 + i] = hi[i]; }
    return a;
}

// B 32x16 fragment where the operand is stored transposed as rows of 32
// contiguous K-elements (weight rows / K rows / V^T rows) with row stride ldr.
// lane<16: col n=lane, K in {0..7,16..23}; lane>=16: col n=lane-16, K in {8..15,24..31}.
static __device__ inline v16bf b_frag_rows(const __bf16* rows, int ldr, int n0, int lane) {
    int n  = lane & 15;
    int kb = (lane >> 4) << 3;
    const __bf16* p = rows + (size_t)(n0 + n) * ldr + kb;
    v8bf lo = *(const v8bf*)(p);
    v8bf hi = *(const v8bf*)(p + 16);
    v16bf b;
#pragma unroll
    for (int i = 0; i < 8; ++i) { b[i] = lo[i]; b[8 + i] = hi[i]; }
    return b;
}

// ---------------------------------------------------------------------------
// f32 -> bf16 convert
// ---------------------------------------------------------------------------
__global__ void cvt_f32_bf16(const float* __restrict__ src, __bf16* __restrict__ dst,
                             size_t n) {
    size_t i = (size_t)blockIdx.x * blockDim.x + threadIdx.x;
    if (i < n) dst[i] = (__bf16)src[i];
}

// ---------------------------------------------------------------------------
// GEMM: C[M,N] = A[M,K] * W[N,K]^T (bf16 in, fp32 out).
// 256 thr / 8 waves, block tile 128x128, K-step 32, double-buffered async LDS.
// Wave tile 32x64. All 6 fragments loaded as one clause, then 8 WMMAs burst.
// ---------------------------------------------------------------------------
__global__ __launch_bounds__(256)
void gemm_bf16_wmma(const __bf16* __restrict__ A, const __bf16* __restrict__ W,
                    float* __restrict__ C, int M, int N, int K, int ldc) {
    __shared__ __align__(64) __bf16 At[2][128 * 32];
    __shared__ __align__(64) __bf16 Bt[2][128 * 32];
    const int tid  = threadIdx.x;
    const int lane = tid & 31;
    const int wave = tid >> 5;
    const int wy   = wave >> 1;            // 0..3 : 32-row band
    const int wx   = wave & 1;             // 0..1 : 64-col band
    const int row0g = blockIdx.x * 128;
    const int col0g = blockIdx.y * 128;

    const int r4 = tid >> 2;               // 0..63 row of 64B line
    const int s4 = (tid & 3) * 16;         // 16B segment in row

    const size_t arow = (size_t)K * 2;     // row pitch in bytes

    // stage one 128x32 A tile + 128x32 B tile (4 async ops per thread)
    auto stage = [&](int buf, int kt) {
        size_t kbyte = (size_t)kt * 64;
#pragma unroll
        for (int i = 0; i < 2; ++i) {
            int row = r4 + i * 64;
            async_copy_b128((unsigned)(size_t)&At[buf][0] + row * 64 + s4,
                            (uint64_t)(size_t)A + (size_t)(row0g + row) * arow + kbyte + s4);
            async_copy_b128((unsigned)(size_t)&Bt[buf][0] + row * 64 + s4,
                            (uint64_t)(size_t)W + (size_t)(col0g + row) * arow + kbyte + s4);
        }
    };

    v8f acc[2][4] = {};
    const int ksteps = K >> 5;

    stage(0, 0);
    for (int kt = 0; kt < ksteps; ++kt) {
        __syncthreads();                            // prior reads of next buf done
        if (kt + 1 < ksteps) {
            stage((kt + 1) & 1, kt + 1);
            asm volatile("s_wait_asynccnt 0x4" ::: "memory");  // current tile landed
        } else {
            asm volatile("s_wait_asynccnt 0x0" ::: "memory");
        }
        __syncthreads();                            // tile visible to all waves

        const __bf16* Ab = &At[kt & 1][0];
        const __bf16* Bb = &Bt[kt & 1][0];
        // hoist ALL fragment loads so WMMAs can issue as one dense burst
        v16bf af[2], bfr[4];
#pragma unroll
        for (int ms = 0; ms < 2; ++ms)
            af[ms] = a_frag_lds(Ab, wy * 32 + ms * 16, lane);
#pragma unroll
        for (int j = 0; j < 4; ++j)
            bfr[j] = b_frag_rows(Bb, 32, wx * 64 + j * 16, lane);
#pragma unroll
        for (int j = 0; j < 4; ++j)
#pragma unroll
            for (int ms = 0; ms < 2; ++ms)
                acc[ms][j] = __builtin_amdgcn_wmma_f32_16x16x32_bf16(
                    false, af[ms], false, bfr[j], (short)0, acc[ms][j], false, false);
    }

    const int cm = (lane >= 16) ? 8 : 0;
    const int cn = lane & 15;
#pragma unroll
    for (int ms = 0; ms < 2; ++ms)
#pragma unroll
        for (int j = 0; j < 4; ++j)
#pragma unroll
            for (int e = 0; e < 8; ++e)
                C[(size_t)(row0g + wy * 32 + ms * 16 + e + cm) * ldc +
                  col0g + wx * 64 + j * 16 + cn] = acc[ms][j][e];
}

// ---------------------------------------------------------------------------
// RoPE (fp32, HF-style) on q/k + bf16 split. V is written TRANSPOSED
// ([NKV*HD, T]) so attention B-fragments become contiguous loads.
// grid (T, 20): y = 0..15 q heads, 16..17 k heads, 18..19 v heads. 64 threads.
// ---------------------------------------------------------------------------
__global__ __launch_bounds__(64)
void rope_split(const float* __restrict__ qkv, const int* __restrict__ pos,
                __bf16* __restrict__ qb, __bf16* __restrict__ kg,
                __bf16* __restrict__ vT) {
    int t  = blockIdx.x;
    int hy = blockIdx.y;
    int i  = threadIdx.x;                            // 0..63 rotation pairs
    const float* row = qkv + (size_t)t * QKV_N;
    if (hy < NH + NKV) {
        const float* src = (hy < NH) ? row + hy * HD
                                     : row + NH * HD + (hy - NH) * HD;
        __bf16* dst = (hy < NH) ? qb + (size_t)t * (NH * HD) + hy * HD
                                : kg + (size_t)t * (NKV * HD) + (hy - NH) * HD;
        float p    = (float)pos[t];
        float invf = __expf(-(float)(2 * i) * (1.0f / (float)HD) * __logf(10000.0f));
        float s, c;
        __sincosf(p * invf, &s, &c);
        float x1 = src[i], x2 = src[i + HD / 2];
        dst[i]          = (__bf16)(x1 * c - x2 * s);
        dst[i + HD / 2] = (__bf16)(x2 * c + x1 * s);
    } else {
        int vh = hy - NH - NKV;
        const float* src = row + (NH + NKV) * HD + vh * HD;
        vT[((size_t)vh * HD + i) * T_TOK + t]          = (__bf16)src[i];
        vT[((size_t)vh * HD + i + HD / 2) * T_TOK + t] = (__bf16)src[i + HD / 2];
    }
}

// ---------------------------------------------------------------------------
// Causal GQA flash attention. Block = 4 waves = one 64-row Q band of one head.
// K tile (32x128) and V^T tile (128x32) double-buffered in LDS via async copy.
// Each wave owns a 16-row Q tile + its softmax state; KV tiles of 32 so the
// P matrix (16x32) is exactly one WMMA A operand, relayed through LDS.
// ---------------------------------------------------------------------------
__global__ __launch_bounds__(128)
void flash_attn(const __bf16* __restrict__ qg, const __bf16* __restrict__ kg,
                const __bf16* __restrict__ vTg, __bf16* __restrict__ attnb) {
    __shared__ __align__(64) __bf16 Kt[2][32 * 128];   // [kv row][feature]
    __shared__ __align__(64) __bf16 Vt[2][128 * 32];   // [feature][kv row]
    __shared__ __align__(64) __bf16 Pl[4][16 * 32];    // per-wave P relay
    const int tid  = threadIdx.x;
    const int lane = tid & 31;
    const int wave = tid >> 5;
    const int h      = blockIdx.x;
    const int q0band = blockIdx.y * 64;
    const int q0     = q0band + wave * 16;
    const int kvh    = h / GROUP;
    const float scale = 0.08838834764831845f;          // 128^-0.5

    // stage one KV tile (8 async ops per thread)
    auto stage = [&](int kv0, int buf) {
#pragma unroll
        for (int i = 0; i < 4; ++i) {                  // K: 32 rows x 256B
            int c = tid + i * 128;
            int r = c >> 4, seg = (c & 15) * 16;
            async_copy_b128((unsigned)(size_t)&Kt[buf][0] + r * 256 + seg,
                            (uint64_t)(size_t)kg +
                                (size_t)(kv0 + r) * (NKV * HD * 2) +
                                (size_t)kvh * HD * 2 + seg);
        }
#pragma unroll
        for (int i = 0; i < 4; ++i) {                  // V^T: 128 rows x 64B
            int c = tid + i * 128;
            int f = c >> 2, seg = (c & 3) * 16;
            async_copy_b128((unsigned)(size_t)&Vt[buf][0] + f * 64 + seg,
                            (uint64_t)(size_t)vTg +
                                ((size_t)(kvh * HD + f) * T_TOK + kv0) * 2 + seg);
        }
    };

    // Q fragments: 16 rows x 128, 4 chunks of K=32 (read once, from global)
    v16bf qf[4];
#pragma unroll
    for (int kc = 0; kc < 4; ++kc)
        qf[kc] = a_frag_global(qg + h * HD, NH * HD, q0, kc * 32, lane);

    const int cm = (lane >= 16) ? 8 : 0;
    const int cn = lane & 15;
    float mst[8], lst[8];
#pragma unroll
    for (int e = 0; e < 8; ++e) { mst[e] = -1e30f; lst[e] = 0.0f; }
    v8f oacc[8] = {};

    const int ntiles = (q0band + 63) / 32 + 1;         // causal bound of band
    stage(0, 0);
    for (int j = 0; j < ntiles; ++j) {
        const int kv0 = j * 32;
        __syncthreads();                                // done reading next buf
        if (j + 1 < ntiles) {
            stage((j + 1) * 32, (j + 1) & 1);
            asm volatile("s_wait_asynccnt 0x8" ::: "memory");
        } else {
            asm volatile("s_wait_asynccnt 0x0" ::: "memory");
        }
        __syncthreads();                                // tile j visible
        const __bf16* Kb = &Kt[j & 1][0];
        const __bf16* Vb = &Vt[j & 1][0];

        // S = Q * K^T : hoist all 8 K fragments, then burst 8 WMMAs
        v8f s[2] = {};
        {
            v16bf bk[8];
#pragma unroll
            for (int nh2 = 0; nh2 < 2; ++nh2)
#pragma unroll
                for (int kc = 0; kc < 4; ++kc)
                    bk[nh2 * 4 + kc] = b_frag_rows(Kb + kc * 32, 128, nh2 * 16, lane);
#pragma unroll
            for (int nh2 = 0; nh2 < 2; ++nh2)
#pragma unroll
                for (int kc = 0; kc < 4; ++kc)
                    s[nh2] = __builtin_amdgcn_wmma_f32_16x16x32_bf16(
                        false, qf[kc], false, bk[nh2 * 4 + kc], (short)0, s[nh2],
                        false, false);
        }
        // scale + causal mask + online softmax (rows = e + cm)
        float corr[8];
#pragma unroll
        for (int e = 0; e < 8; ++e) {
            int rowq = q0 + e + cm;
            float v0 = s[0][e] * scale;
            float v1 = s[1][e] * scale;
            if (kv0 + cn      > rowq) v0 = -1e30f;
            if (kv0 + 16 + cn > rowq) v1 = -1e30f;
            float rm = fmaxf(v0, v1);
#pragma unroll
            for (int off = 1; off < 16; off <<= 1)
                rm = fmaxf(rm, __shfl_xor(rm, off, 32));
            float mn = fmaxf(mst[e], rm);
            corr[e]  = __expf(mst[e] - mn);
            mst[e]   = mn;
            float p0 = __expf(v0 - mn);
            float p1 = __expf(v1 - mn);
            Pl[wave][(e + cm) * 32 + cn]      = (__bf16)p0;
            Pl[wave][(e + cm) * 32 + 16 + cn] = (__bf16)p1;
            float rs = p0 + p1;
#pragma unroll
            for (int off = 1; off < 16; off <<= 1)
                rs += __shfl_xor(rs, off, 32);
            lst[e] = lst[e] * corr[e] + rs;
        }
        asm volatile("s_wait_dscnt 0x0" ::: "memory");  // P relay landed (own wave)
        // P fragment (A layout) from LDS relay
        v16bf pf;
        {
            int am  = lane & 15;
            int akb = (lane >> 4) << 3;
#pragma unroll
            for (int i = 0; i < 8; ++i) {
                pf[i]     = Pl[wave][am * 32 + akb + i];
                pf[8 + i] = Pl[wave][am * 32 + 16 + akb + i];
            }
        }
        // O = diag(corr)*O + P * V : hoist V fragments, then burst WMMAs
        {
            v16bf bv[8];
#pragma unroll
            for (int oc = 0; oc < 8; ++oc)
                bv[oc] = b_frag_rows(Vb, 32, oc * 16, lane);
#pragma unroll
            for (int oc = 0; oc < 8; ++oc) {
                v8f c;
#pragma unroll
                for (int e = 0; e < 8; ++e) c[e] = oacc[oc][e] * corr[e];
                oacc[oc] = __builtin_amdgcn_wmma_f32_16x16x32_bf16(
                    false, pf, false, bv[oc], (short)0, c, false, false);
            }
        }
    }
    // normalize + store bf16 into attn [T, NH*HD]
#pragma unroll
    for (int oc = 0; oc < 8; ++oc)
#pragma unroll
        for (int e = 0; e < 8; ++e) {
            float o = oacc[oc][e] / lst[e];
            attnb[(size_t)(q0 + e + cm) * (NH * HD) + h * HD + oc * 16 + cn] =
                (__bf16)o;
        }
}

// ---------------------------------------------------------------------------
extern "C" void kernel_launch(void* const* d_in, const int* in_sizes, int n_in,
                              void* d_out, int out_size, void* d_ws, size_t ws_size,
                              hipStream_t stream) {
    const float* hs = (const float*)d_in[0];
    const int*   ps = (const int*)d_in[1];
    const float* Wq = (const float*)d_in[2];
    const float* Wk = (const float*)d_in[3];
    const float* Wv = (const float*)d_in[4];
    const float* Wo = (const float*)d_in[5];
    float* out = (float*)d_out;

    char* ws = (char*)d_ws;
    size_t off = 0;
    __bf16* hsb  = (__bf16*)(ws + off); off += (size_t)T_TOK * HIDN  * 2;
    __bf16* wqkv = (__bf16*)(ws + off); off += (size_t)QKV_N * HIDN  * 2;
    __bf16* wob  = (__bf16*)(ws + off); off += (size_t)HIDN  * HIDN  * 2;
    float*  qkvf = (float* )(ws + off); off += (size_t)T_TOK * QKV_N * 4;
    __bf16* qb   = (__bf16*)(ws + off); off += (size_t)T_TOK * NH  * HD * 2;
    __bf16* kb   = (__bf16*)(ws + off); off += (size_t)T_TOK * NKV * HD * 2;
    __bf16* vT   = (__bf16*)(ws + off); off += (size_t)T_TOK * NKV * HD * 2;
    __bf16* attnb= (__bf16*)(ws + off); off += (size_t)T_TOK * NH  * HD * 2;

    {   // bf16 conversions (hidden + fused [Wq;Wk;Wv] + Wo)
        size_t n;
        n = (size_t)T_TOK * HIDN;
        cvt_f32_bf16<<<dim3((unsigned)((n + 255) / 256)), 256, 0, stream>>>(hs, hsb, n);
        n = (size_t)NH * HD * HIDN;
        cvt_f32_bf16<<<dim3((unsigned)((n + 255) / 256)), 256, 0, stream>>>(Wq, wqkv, n);
        n = (size_t)NKV * HD * HIDN;
        cvt_f32_bf16<<<dim3((unsigned)((n + 255) / 256)), 256, 0, stream>>>(
            Wk, wqkv + (size_t)NH * HD * HIDN, n);
        cvt_f32_bf16<<<dim3((unsigned)((n + 255) / 256)), 256, 0, stream>>>(
            Wv, wqkv + (size_t)(NH + NKV) * HD * HIDN, n);
        n = (size_t)HIDN * HIDN;
        cvt_f32_bf16<<<dim3((unsigned)((n + 255) / 256)), 256, 0, stream>>>(Wo, wob, n);
    }

    // QKV projection: [T,2560] = hs[T,2048] x Wqkv[2560,2048]^T
    gemm_bf16_wmma<<<dim3(T_TOK / 128, QKV_N / 128), 256, 0, stream>>>(
        hsb, wqkv, qkvf, T_TOK, QKV_N, HIDN, QKV_N);

    // fp32 RoPE + split to bf16 q/k/vT
    rope_split<<<dim3(T_TOK, NH + 2 * NKV), 64, 0, stream>>>(qkvf, ps, qb, kb, vT);

    // causal GQA flash attention (per head, 64-row Q bands)
    flash_attn<<<dim3(NH, T_TOK / 64), 128, 0, stream>>>(qb, kb, vT, attnb);

    // output projection: out[T,2048] = attn[T,2048] x Wo[2048,2048]^T
    gemm_bf16_wmma<<<dim3(T_TOK / 128, HIDN / 128), 256, 0, stream>>>(
        attnb, wob, out, T_TOK, HIDN, NH * HD, HIDN);
}
